// SynthesisNetwork_62594853372216
// MI455X (gfx1250) — matrix operand order repaired and checked by reading, not compiled, not run on hardware
//
#include <hip/hip_runtime.h>
#include <math.h>

typedef _Float16 v16h __attribute__((ext_vector_type(16)));
typedef _Float16 v8h  __attribute__((ext_vector_type(8)));
typedef float    v8f  __attribute__((ext_vector_type(8)));

#define B_N        16
#define WDIM       512
#define FREQ_N     1024
#define CH_N       512
#define OCH_N      512
#define HW_N       4096
#define KSTEPS_FEAT 32     // 1024/32
#define KSTEPS_CONV 144    // 4608/32
#define GAIN_512   0.04419417382415922f   // 1/sqrt(512)

// ---- fragment index helpers (CDNA5 ISA 7.12.2 / 7.12.4 wave32 layouts) ----
// A (f16 16x32): lanes 0-15 row M=lane, K = {0..7,16..23}; lanes 16-31 same M, K = {8..15,24..31}
__device__ __forceinline__ int a_k_index(int lane, int hh) {
  return ((lane & 16) ? 8 : 0) + (hh < 8 ? hh : hh + 8);
}
// B (f16 32x16): lanes 0-15 col N=lane, K = {0..7,16..23}; lanes 16-31 same N, K = {8..15,24..31}
__device__ __forceinline__ int b_k_index(int lane, int hh) {
  return ((lane & 16) ? 8 : 0) + (hh < 8 ? hh : hh + 8);
}

// ---- gfx1250 async global->LDS copy (ASYNCcnt), per CDNA5 ISA ----
__device__ __forceinline__ void async_copy_b128(unsigned lds_off, const void* gptr) {
  asm volatile("global_load_async_to_lds_b128 %0, %1, off"
               :: "v"(lds_off), "v"((unsigned long long)(uintptr_t)gptr)
               : "memory");
}
__device__ __forceinline__ void wait_async0() {
  asm volatile("s_wait_asynccnt 0x0" ::: "memory");
}

// ---------------------------------------------------------------- geometry
__global__ void k_geom(const float* __restrict__ w, const float* __restrict__ affine_w,
                       const float* __restrict__ affine_b, const float* __restrict__ freqs,
                       const float* __restrict__ phases, float* __restrict__ freq4) {
  __shared__ float t[4];
  __shared__ float prm[4];
  int b = blockIdx.x, tid = threadIdx.x;
  if (tid < 4) {
    float acc = 0.f;
    for (int k = 0; k < WDIM; ++k) acc = fmaf(w[b*WDIM + k], affine_w[tid*WDIM + k], acc);
    t[tid] = acc * GAIN_512 + affine_b[tid];
  }
  __syncthreads();
  if (tid == 0) {
    float rn = rsqrtf(t[0]*t[0] + t[1]*t[1]);
    float c = t[0]*rn, s = t[1]*rn, tx = t[2]*rn, ty = t[3]*rn;
    prm[0] = c; prm[1] = s;
    prm[2] = -c*tx + s*ty;       // shift = rot @ (-tx,-ty)
    prm[3] = -s*tx - c*ty;
  }
  __syncthreads();
  float c = prm[0], s = prm[1], sh0 = prm[2], sh1 = prm[3];
  for (int f = tid; f < FREQ_N; f += blockDim.x) {
    float f0 = freqs[2*f], f1 = freqs[2*f+1];
    float fb0 = f0*c + f1*s;
    float fb1 = -f0*s + f1*c;
    float phb = phases[f] + f0*sh0 + f1*sh1;
    float nrm = sqrtf(fb0*fb0 + fb1*fb1);
    float amp = 1.0f - (nrm - 2.0f) * (1.0f/30.0f);   // (|f|-BW)/(SR/2-BW)
    amp = fminf(fmaxf(amp, 0.0f), 1.0f);
    float4 q; q.x = fb0; q.y = fb1; q.z = phb; q.w = amp;
    ((float4*)freq4)[b*FREQ_N + f] = q;
  }
}

// ---------------------------------------------------------------- style FC
__global__ void k_style(const float* __restrict__ w, const float* __restrict__ style_w,
                        const float* __restrict__ style_b, float* __restrict__ s_arr) {
  int g = blockIdx.x * 256 + threadIdx.x;
  int b = g >> 9, c = g & 511;
  const float* wp = w + b*WDIM;
  const float* sw = style_w + (size_t)c*WDIM;
  float acc = 0.f;
  for (int k = 0; k < WDIM; ++k) acc = fmaf(wp[k], sw[k], acc);
  s_arr[g] = acc * GAIN_512 + style_b[c];
}

// --------------------------------------------- global scalars: s_scale, ef_scale
__global__ void k_scalars(const float* __restrict__ s_arr, const float* __restrict__ freqs,
                          float* __restrict__ scalars) {
  __shared__ float red[256];
  __shared__ float keep;
  int tid = threadIdx.x;
  float ss = 0.f;
  for (int i = tid; i < B_N*CH_N; i += 256) { float v = s_arr[i]; ss += v*v; }
  red[tid] = ss; __syncthreads();
  for (int o = 128; o > 0; o >>= 1) { if (tid < o) red[tid] += red[tid+o]; __syncthreads(); }
  if (tid == 0) keep = red[0];
  __syncthreads();
  float cnt = 0.f;
  for (int f = tid; f < FREQ_N; f += 256) {
    float f0 = freqs[2*f], f1 = freqs[2*f+1];
    if (sqrtf(f0*f0 + f1*f1) < 2.0f) cnt += 1.f;
  }
  red[tid] = cnt; __syncthreads();
  for (int o = 128; o > 0; o >>= 1) { if (tid < o) red[tid] += red[tid+o]; __syncthreads(); }
  if (tid == 0) {
    scalars[0] = rsqrtf(keep / (float)(B_N*CH_N));   // s_hat scale
    scalars[1] = rsqrtf(red[0]);                     // 1/sqrt(effect_freq)
  }
}

// ------------------------------------------- conv weight RMS norm + per-(o,i) sq
__global__ void k_wnorm(const float* __restrict__ conv_w, float* __restrict__ wsq,
                        float* __restrict__ wnorm) {
  int o = blockIdx.x, tid = threadIdx.x;
  __shared__ float red[256];
  float tot = 0.f;
  for (int ci = tid; ci < CH_N; ci += 256) {
    const float* p = conv_w + ((size_t)o*CH_N + ci)*9;
    float s9 = 0.f;
    #pragma unroll
    for (int k = 0; k < 9; ++k) s9 = fmaf(p[k], p[k], s9);
    wsq[o*CH_N + ci] = s9;
    tot += s9;
  }
  red[tid] = tot; __syncthreads();
  for (int off = 128; off > 0; off >>= 1) { if (tid < off) red[tid] += red[tid+off]; __syncthreads(); }
  if (tid == 0) wnorm[o] = rsqrtf(red[0] / (float)(CH_N*9));
}

// ---------------------------------------------------------------- demod coefs
__global__ void k_dcoef(const float* __restrict__ s_arr, const float* __restrict__ scalars,
                        const float* __restrict__ wsq, const float* __restrict__ wnorm,
                        float* __restrict__ dcoef) {
  int g = blockIdx.x * 256 + threadIdx.x;
  int b = g >> 9, o = g & 511;
  float s_scale = scalars[0];
  float acc = 0.f;
  for (int i = 0; i < CH_N; ++i) {
    float sh = s_arr[b*CH_N + i] * s_scale;
    acc = fmaf(wsq[o*CH_N + i], sh*sh, acc);
  }
  float wn = wnorm[o];
  dcoef[g] = rsqrtf(wn*wn*acc + 1e-8f);
}

// --------------------------- prepack input_weight into B-fragment order (f16)
__global__ void k_pack_iw(const float* __restrict__ input_weight,
                          const float* __restrict__ scalars, _Float16* __restrict__ iwpk) {
  int id = blockIdx.x * 256 + threadIdx.x;          // 32*32*32*16 = 524288 exact
  int hh = id & 15, lane = (id >> 4) & 31, kstep = (id >> 9) & 31, nsub = id >> 14;
  int c = nsub*16 + (lane & 15);
  int K = kstep*32 + b_k_index(lane, hh);
  iwpk[id] = (_Float16)(input_weight[(size_t)c*FREQ_N + K] * scalars[1]);
}

// --------------------------- prepack w_hat into B-fragment order (f16)
__global__ void k_pack_w(const float* __restrict__ conv_w, const float* __restrict__ wnorm,
                         _Float16* __restrict__ wpk) {
  int id = blockIdx.x * 256 + threadIdx.x;          // 32*144*32*16 = 2359296 exact
  int hh = id & 15, lane = (id >> 4) & 31;
  int t = id >> 9;
  int kstep = t % KSTEPS_CONV, nsub = t / KSTEPS_CONV;
  int co = nsub*16 + (lane & 15);
  int K = kstep*32 + b_k_index(lane, hh);
  int blk = K >> 9, ci = K & 511;
  int kh = blk / 3, kw = blk % 3;
  float v = conv_w[(((size_t)co*CH_N + ci)*3 + kh)*3 + kw] * wnorm[co];
  wpk[id] = (_Float16)v;
}

// ------------- fused sinusoid synthesis + feature GEMM + style modulation -> NHWC f16
// Block tile 64(M) x 256(N): 8 waves as 4Mx2N, wave tile 16x128 (8 accumulators).
// N-grid = 2 -> each sinusoid generated only 2x (vs 8x at N=64).
__global__ __launch_bounds__(256) void k_feat_gemm(const float* __restrict__ freq4,
    const float* __restrict__ s_arr, const float* __restrict__ scalars,
    const _Float16* __restrict__ iwpk, _Float16* __restrict__ xmod) {
  __shared__ float4 shf[FREQ_N];                    // 16 KB: (fb0,fb1,ph,amp) per freq
  int tid = threadIdx.x;
  int b = blockIdx.z;
  for (int i = tid; i < FREQ_N; i += 256) shf[i] = ((const float4*)freq4)[b*FREQ_N + i];
  __syncthreads();

  int lane = tid & 31, wave = tid >> 5;
  int waveM = wave & 3, waveN = wave >> 2;
  int m_base = blockIdx.x*64 + waveM*16;
  int row = m_base + (lane & 15);
  int h = row >> 6, wc = row & 63;
  float gx = ((wc + 0.5f)*(2.0f/64.0f) - 1.0f)*0.5f;    // theta00 = 0.5
  float gy = ((h  + 0.5f)*(2.0f/64.0f) - 1.0f)*0.5f;
  const float TWOPI = 6.28318530717958647f;

  int nsub0 = blockIdx.y*16 + waveN*8;
  v8f acc[8] = {};
  for (int kstep = 0; kstep < KSTEPS_FEAT; ++kstep) {
    v16h a;
    #pragma unroll
    for (int hh = 0; hh < 16; ++hh) {
      int f = kstep*32 + a_k_index(lane, hh);
      float4 q = shf[f];
      float arg = fmaf(gx, q.x, fmaf(gy, q.y, q.z));
      a[hh] = (_Float16)(__sinf(TWOPI * arg) * q.w);
    }
    const _Float16* bp = iwpk + ((((size_t)nsub0*KSTEPS_FEAT + kstep)*32 + lane) << 4);
    #pragma unroll
    for (int j = 0; j < 8; ++j) {
      v16h bf = *(const v16h*)(bp + (size_t)j*16384);   // nsub stride = 32*32*16
      acc[j] = __builtin_amdgcn_wmma_f32_16x16x32_f16(false, a, false, bf, (short)0, acc[j], false, false);
    }
  }

  float s_scale = scalars[0];
  int m0 = m_base + ((lane & 16) ? 8 : 0);
  #pragma unroll
  for (int j = 0; j < 8; ++j) {
    int c = blockIdx.y*256 + waveN*128 + j*16 + (lane & 15);
    float sh = s_arr[b*CH_N + c] * s_scale;
    _Float16* dst = xmod + ((size_t)b*HW_N + m0)*CH_N + c;
    #pragma unroll
    for (int r = 0; r < 8; ++r) dst[(size_t)r*CH_N] = (_Float16)(acc[j][r] * sh);
  }
}

// -------------------- implicit-GEMM 3x3 conv, LDS-staged via async global->LDS
// Block tile 128(M = 2 image rows) x 64(N). Outer loop: 16 ci-chunks of 32 channels;
// each chunk stages a 4row x 64col x 32ch halo tile (16 KB) into LDS with
// global_load_async_to_lds_b128 (double-buffered), then runs the 9 conv taps
// (36 WMMAs) out of LDS. Global A traffic /9, fully bursted.
__global__ __launch_bounds__(256) void k_conv_gemm(const _Float16* __restrict__ xmod,
    const _Float16* __restrict__ wpk, const float* __restrict__ dcoef,
    const float* __restrict__ conv_b, float* __restrict__ out) {
  __shared__ __align__(16) _Float16 tile[2][4*64*32];   // 2 x 16 KB
  int tid = threadIdx.x, lane = tid & 31, wave = tid >> 5;
  int b = blockIdx.z;
  int h0 = blockIdx.x * 2;                      // output rows h0, h0+1
  int m_base = blockIdx.x*128 + wave*16;
  int row = m_base + (lane & 15);
  int h = row >> 6, wc = row & 63;
  int cb_lane = (lane & 16) ? 8 : 0;
  unsigned lds_base = (unsigned)(uintptr_t)&tile[0][0];   // low 32 bits = LDS offset

  const _Float16* gB = xmod + (size_t)b*HW_N*CH_N;

  // stage chunk `ci_chunk` into buffer `buf`: 1024 16B-chunks, 4 per thread
  auto issue_chunk = [&](int ci_chunk, int buf) {
    const _Float16* gbase = gB + ci_chunk*32;
    #pragma unroll
    for (int k = 0; k < 4; ++k) {
      int c = k*256 + tid;                      // 0..1023
      int r = c >> 8, col = (c >> 2) & 63, s = c & 3;
      int rr = h0 - 1 + r;                      // clamp OOB rows (never read back)
      rr = rr < 0 ? 0 : (rr > 63 ? 63 : rr);
      const _Float16* g = gbase + ((size_t)(rr*64 + col))*CH_N + s*8;
      async_copy_b128(lds_base + (unsigned)(buf*32768 + c*16), g);
    }
  };

  v8f acc0 = {}, acc1 = {}, acc2 = {}, acc3 = {};
  issue_chunk(0, 0);
  for (int chunk = 0; chunk < 16; ++chunk) {
    int buf = chunk & 1;
    wait_async0();             // own wave's DMA into LDS complete
    __syncthreads();           // all waves' DMA complete; prev compute done
    if (chunk + 1 < 16) issue_chunk(chunk + 1, buf ^ 1);   // overlap next DMA

    const _Float16* tbase = &tile[buf][0];
    #pragma unroll
    for (int blk = 0; blk < 9; ++blk) {
      int kh = blk / 3, kw = blk % 3;
      int hi = h + kh - 1, wi = wc + kw - 1;
      int r = h - h0 + kh;                      // tile row 0..3
      v16h a = {};
      if ((unsigned)hi < 64u && (unsigned)wi < 64u) {
        const _Float16* src = tbase + (r*64 + wi)*32 + cb_lane;
        v8h lo  = *(const v8h*)src;             // ci = cb .. cb+7
        v8h hi8 = *(const v8h*)(src + 16);      // ci = cb+16 .. cb+23
        #pragma unroll
        for (int i = 0; i < 8; ++i) { a[i] = lo[i]; a[8+i] = hi8[i]; }
      }
      int kstep = blk*16 + chunk;               // K-order matches wpk packing
      const _Float16* bp = wpk + ((((size_t)(blockIdx.y*4)*KSTEPS_CONV + kstep)*32 + lane) << 4);
      v16h b0 = *(const v16h*)(bp);
      v16h b1 = *(const v16h*)(bp + 73728);     // nsub stride = 144*32*16
      v16h b2 = *(const v16h*)(bp + 147456);
      v16h b3 = *(const v16h*)(bp + 221184);
      acc0 = __builtin_amdgcn_wmma_f32_16x16x32_f16(false, a, false, b0, (short)0, acc0, false, false);
      acc1 = __builtin_amdgcn_wmma_f32_16x16x32_f16(false, a, false, b1, (short)0, acc1, false, false);
      acc2 = __builtin_amdgcn_wmma_f32_16x16x32_f16(false, a, false, b2, (short)0, acc2, false, false);
      acc3 = __builtin_amdgcn_wmma_f32_16x16x32_f16(false, a, false, b3, (short)0, acc3, false, false);
    }
  }

  int m0 = m_base + ((lane & 16) ? 8 : 0);
  v8f accs[4] = {acc0, acc1, acc2, acc3};
  const float SQRT2 = 1.41421356237309505f;
  #pragma unroll
  for (int j = 0; j < 4; ++j) {
    int co = blockIdx.y*64 + j*16 + (lane & 15);
    float d = dcoef[b*OCH_N + co];
    float bias = conv_b[co];
    float* dst = out + ((size_t)b*OCH_N + co)*HW_N + m0;
    #pragma unroll
    for (int r = 0; r < 8; ++r) {
      float v = fmaf(accs[j][r], d, bias);
      v = (v >= 0.f ? v : 0.2f*v) * SQRT2;
      v = fminf(fmaxf(v, -256.f), 256.f);
      dst[r] = v;
    }
  }
}

// ----------------------------------------------------------------- launcher
extern "C" void kernel_launch(void* const* d_in, const int* in_sizes, int n_in,
                              void* d_out, int out_size, void* d_ws, size_t ws_size,
                              hipStream_t stream) {
  (void)in_sizes; (void)n_in; (void)out_size; (void)ws_size;
  const float* w        = (const float*)d_in[0];
  const float* affine_w = (const float*)d_in[1];
  const float* affine_b = (const float*)d_in[2];
  const float* freqs    = (const float*)d_in[3];
  const float* phases   = (const float*)d_in[4];
  const float* input_w  = (const float*)d_in[5];
  const float* style_w  = (const float*)d_in[6];
  const float* style_b  = (const float*)d_in[7];
  const float* conv_w   = (const float*)d_in[8];
  const float* conv_b   = (const float*)d_in[9];

  char* ws = (char*)d_ws;                       // all offsets 256B aligned
  float*    freq4   = (float*)   (ws + 0);        //  256 KB  (B*1024 float4)
  float*    s_arr   = (float*)   (ws + 262144);   //   32 KB
  float*    scalars = (float*)   (ws + 294912);   //  s_scale, ef_scale
  float*    wnorm   = (float*)   (ws + 295168);   //    2 KB
  float*    wsq     = (float*)   (ws + 297216);   //    1 MB
  float*    dcoef   = (float*)   (ws + 1345792);  //   32 KB
  _Float16* iwpk    = (_Float16*)(ws + 1378560);  //    1 MB  fragment-packed
  _Float16* wpk     = (_Float16*)(ws + 2427136);  //  4.5 MB  fragment-packed
  _Float16* xmod    = (_Float16*)(ws + 7145728);  //   64 MB  NHWC f16 (L2-resident)
  float* out = (float*)d_out;

  k_geom   <<<B_N, 256, 0, stream>>>(w, affine_w, affine_b, freqs, phases, freq4);
  k_style  <<<32, 256, 0, stream>>>(w, style_w, style_b, s_arr);
  k_scalars<<<1, 256, 0, stream>>>(s_arr, freqs, scalars);
  k_wnorm  <<<OCH_N, 256, 0, stream>>>(conv_w, wsq, wnorm);
  k_dcoef  <<<32, 256, 0, stream>>>(s_arr, scalars, wsq, wnorm, dcoef);
  k_pack_iw<<<2048, 256, 0, stream>>>(input_w, scalars, iwpk);
  k_pack_w <<<9216, 256, 0, stream>>>(conv_w, wnorm, wpk);

  dim3 gf(64, 2, B_N);   // M-tiles(64) x N-tiles(256) x batch
  k_feat_gemm<<<gf, 256, 0, stream>>>(freq4, s_arr, scalars, iwpk, xmod);
  dim3 gc(32, 8, B_N);   // M-tiles(128 = 2 rows) x N-tiles(64) x batch
  k_conv_gemm<<<gc, 256, 0, stream>>>(xmod, wpk, dcoef, conv_b, out);
}
